// QuantOPTAttention_55817394978968
// MI455X (gfx1250) — 16-bit and 8-bit matrix operand order repaired and checked by reading, compiled, not run on hardware
//
#include <hip/hip_runtime.h>


typedef __attribute__((ext_vector_type(16))) _Float16 v16h;
typedef __attribute__((ext_vector_type(8)))  _Float16 v8h;
typedef __attribute__((ext_vector_type(8)))  float    v8f;
typedef __attribute__((ext_vector_type(8)))  int      v8i;
typedef __attribute__((ext_vector_type(4)))  int      v4i;

#define NEGF (-3.4028234663852886e38f)

#define CB   2
#define CT   1024
#define CE   2048
#define CH   32
#define CD   64
#define CBT  (CB * CT)
#define CBH  (CB * CH)

#if defined(__HIP_DEVICE_COMPILE__) && __has_builtin(__builtin_amdgcn_global_load_async_to_lds_b128)
#define ASYNC_LDS 1
#else
#define ASYNC_LDS 0
#endif

#if ASYNC_LDS
typedef __attribute__((address_space(1))) v4i* gptr_v4i;
typedef __attribute__((address_space(3))) v4i* lptr_v4i;
__device__ __forceinline__ void async_copy16(const signed char* g, signed char* l) {
  __builtin_amdgcn_global_load_async_to_lds_b128(
      (gptr_v4i)(g), (lptr_v4i)(l), 0, 0);
}
__device__ __forceinline__ void async_wait_all() {
#if __has_builtin(__builtin_amdgcn_s_wait_asynccnt)
  __builtin_amdgcn_s_wait_asynccnt(0);
#else
  asm volatile("s_wait_asynccnt 0x0" ::: "memory");
#endif
}
#endif

__device__ __forceinline__ float waveMax32(float v) {
#pragma unroll
  for (int o = 16; o; o >>= 1) v = fmaxf(v, __shfl_xor(v, o, 32));
  return v;
}
__device__ __forceinline__ float waveSum32(float v) {
#pragma unroll
  for (int o = 16; o; o >>= 1) v += __shfl_xor(v, o, 32);
  return v;
}
__device__ __forceinline__ float blockMax256(float v) {
  __shared__ float red[8];
  v = waveMax32(v);
  if ((threadIdx.x & 31) == 0) red[threadIdx.x >> 5] = v;
  __syncthreads();
  if (threadIdx.x == 0) {
    float m = red[0];
#pragma unroll
    for (int i = 1; i < 8; i++) m = fmaxf(m, red[i]);
    red[0] = m;
  }
  __syncthreads();
  float r = red[0];
  __syncthreads();
  return r;
}

__global__ void k_zero(float* acc, int* flags, unsigned* vmax) {
  for (int i = threadIdx.x; i < CT; i += 256) { acc[i] = 0.f; flags[i] = 0; }
  if (threadIdx.x == 0) *vmax = 0u;
}

__global__ void k_rowquant8(const float* __restrict__ X, signed char* __restrict__ Xq,
                            float* __restrict__ S, int C) {
  const int row = blockIdx.x;
  const float* x = X + (size_t)row * C;
  float m = 0.f;
  for (int c = threadIdx.x; c < C; c += 256) m = fmaxf(m, fabsf(x[c]));
  m = blockMax256(m);
  const float s = fmaxf(m, 1e-5f) / 127.f;
  if (threadIdx.x == 0) S[row] = s;
  const float inv = 1.f / s;
  signed char* q = Xq + (size_t)row * C;
  for (int c = threadIdx.x; c < C; c += 256) q[c] = (signed char)lrintf(x[c] * inv);
}

__global__ void k_post_q(const float* __restrict__ Y, _Float16* __restrict__ Qh) {
  const int row = blockIdx.x;
  const int b = row >> 10, t = row & 1023;
  const float* y = Y + (size_t)row * CE;
  float m = 0.f;
  for (int c = threadIdx.x; c < CE; c += 256) m = fmaxf(m, fabsf(y[c]));
  m = blockMax256(m);
  const float s = fmaxf(m, 1e-5f) / 127.f;
  for (int c = threadIdx.x; c < CE; c += 256) {
    const float v = rintf(y[c] / s) * s * 0.125f;
    const int h = c >> 6, d = c & 63;
    Qh[(((size_t)(b * CH + h)) * CT + t) * CD + d] = (_Float16)v;
  }
}

__global__ void k_post_kv(const float* __restrict__ Y, _Float16* __restrict__ Kh) {
  const int row = blockIdx.x;
  const int b = row >> 10, t = row & 1023;
  const float* y = Y + (size_t)row * CE;
  for (int c = threadIdx.x; c < CE; c += 256) {
    const int h = c >> 6, d = c & 63;
    Kh[(((size_t)(b * CH + h)) * CT + t) * CD + d] = (_Float16)y[c];
  }
}

__global__ void k_absmax(const float* __restrict__ X, int n, unsigned* out) {
  float m = 0.f;
  for (int i = blockIdx.x * 256 + threadIdx.x; i < n; i += gridDim.x * 256)
    m = fmaxf(m, fabsf(X[i]));
  m = blockMax256(m);
  if (threadIdx.x == 0) atomicMax(out, __float_as_uint(m));
}

__global__ void k_topk(const float* __restrict__ acc, int* __restrict__ flags) {
  __shared__ float v[256];
  const int g = blockIdx.x, i = threadIdx.x, pos = g * 256 + i;
  const float a = acc[pos] / (float)(CT - pos) / (float)CBH;
  v[i] = a;
  __syncthreads();
  int rank = 0;
  for (int u = 0; u < 256; u++) {
    const float b = v[u];
    rank += (b > a) || (b == a && u < i);
  }
  if (rank < 128) flags[pos] = 1;
}

__global__ void k_requant(const float* __restrict__ K, const float* __restrict__ V,
                          const int* __restrict__ flags, const unsigned* __restrict__ vmax,
                          _Float16* __restrict__ Kq, _Float16* __restrict__ VqT) {
  const int row = blockIdx.x;
  const int b = row >> 10, t = row & 1023;
  const float* kr = K + (size_t)row * CE;
  const float* vr = V + (size_t)row * CE;
  float m = 0.f;
  for (int c = threadIdx.x; c < CE; c += 256) m = fmaxf(m, fabsf(kr[c]));
  m = blockMax256(m);
  const float sk = flags[t] ? fmaxf(m, 1e-5f) / 127.f : fmaxf(m, 1e-5f) / 7.f;
  const float sv = fmaxf(__uint_as_float(*vmax), 1e-5f) / 7.f;
  for (int c = threadIdx.x; c < CE; c += 256) {
    const int h = c >> 6, d = c & 63;
    const size_t bh = (size_t)(b * CH + h);
    Kq[(bh * CT + t) * CD + d]  = (_Float16)(rintf(kr[c] / sk) * sk);
    VqT[(bh * CD + d) * CT + t] = (_Float16)(rintf(vr[c] / sv) * sv);
  }
}

__global__ void k_gemm_i8(const signed char* __restrict__ A, const float* __restrict__ sa,
                          const signed char* __restrict__ Bm, const float* __restrict__ sb,
                          const float* __restrict__ bias, float* __restrict__ C,
                          int M, int N, int K) {
  __shared__ __align__(16) signed char At[2][128 * 64];
  __shared__ __align__(16) signed char Bt[2][128 * 64];
  const int tid = threadIdx.x;
  const int wv = tid >> 5, l = tid & 31, lg = l >> 4, ln = l & 15;
  const int m0 = blockIdx.y * 128, n0 = blockIdx.x * 128;
  const int mW = (wv & 3) * 32, nW = (wv >> 2) * 64;

  v8i acc[2][4];
  const v8i zi = {0, 0, 0, 0, 0, 0, 0, 0};
#pragma unroll
  for (int i = 0; i < 2; i++)
#pragma unroll
    for (int j = 0; j < 4; j++) acc[i][j] = zi;

  auto stage = [&](int kb, int buf) {
#pragma unroll
    for (int c = tid; c < 512; c += 256) {
      const int r = c >> 2, o = (c & 3) * 16;
      const signed char* ga = A  + (size_t)(m0 + r) * K + kb + o;
      const signed char* gb = Bm + (size_t)(n0 + r) * K + kb + o;
#if ASYNC_LDS
      async_copy16(ga, At[buf] + r * 64 + o);
      async_copy16(gb, Bt[buf] + r * 64 + o);
#else
      *(int4*)(At[buf] + r * 64 + o) = *(const int4*)ga;
      *(int4*)(Bt[buf] + r * 64 + o) = *(const int4*)gb;
#endif
    }
  };

  stage(0, 0);
#if ASYNC_LDS
  async_wait_all();
#endif
  __syncthreads();

  int cur = 0;
  for (int kb = 0; kb < K; kb += 64) {
    if (kb + 64 < K) stage(kb + 64, cur ^ 1);

    v8i af[2];
#pragma unroll
    for (int tm = 0; tm < 2; tm++) {
      const signed char* ar = At[cur] + (mW + tm * 16 + ln) * 64;
      const int4 c0 = *(const int4*)(ar +  0 + lg * 16);
      const int4 c1 = *(const int4*)(ar + 32 + lg * 16);
      v8i a;
      a[0] = c0.x; a[1] = c0.y; a[2] = c0.z; a[3] = c0.w;
      a[4] = c1.x; a[5] = c1.y; a[6] = c1.z; a[7] = c1.w;
      af[tm] = a;
    }
    v8i bf[4];
#pragma unroll
    for (int tn = 0; tn < 4; tn++) {
      const signed char* br = Bt[cur] + (nW + tn * 16 + ln) * 64;
      const int4 c0 = *(const int4*)(br +  0 + lg * 16);
      const int4 c1 = *(const int4*)(br + 32 + lg * 16);
      v8i b;
      b[0] = c0.x; b[1] = c0.y; b[2] = c0.z; b[3] = c0.w;
      b[4] = c1.x; b[5] = c1.y; b[6] = c1.z; b[7] = c1.w;
      bf[tn] = b;
    }
#pragma unroll
    for (int tm = 0; tm < 2; tm++)
#pragma unroll
      for (int tn = 0; tn < 4; tn++)
        acc[tm][tn] = __builtin_amdgcn_wmma_i32_16x16x64_iu8(
            true, af[tm], true, bf[tn], acc[tm][tn], false, false);

#if ASYNC_LDS
    async_wait_all();
#endif
    __syncthreads();
    cur ^= 1;
  }

#pragma unroll
  for (int tm = 0; tm < 2; tm++)
#pragma unroll
    for (int tn = 0; tn < 4; tn++) {
      const int n = n0 + nW + tn * 16 + ln;
      const float swn = sb[n], bn = bias[n];
#pragma unroll
      for (int e = 0; e < 8; e++) {
        const int m = m0 + mW + tm * 16 + e + lg * 8;
        C[(size_t)m * N + n] = (float)acc[tm][tn][e] * sa[m] * swn + bn;
      }
    }
}

template <bool PASS1>
__global__ void k_attn(const _Float16* __restrict__ Q, const _Float16* __restrict__ Kh,
                       const _Float16* __restrict__ VT, float* __restrict__ acc,
                       float* __restrict__ ctx) {
  __shared__ float    sc[16][1024];
  __shared__ _Float16 pr[16][1024];
  const int bh = blockIdx.y;
  const int tBase = blockIdx.x * 16;
  const int tid = threadIdx.x;
  const int w = tid >> 5, l = tid & 31, lg = l >> 4, ln = l & 15;

  const _Float16* qrow = Q + (((size_t)bh * CT) + tBase + ln) * CD;
  v16h a0, a1;
  {
    const v8h c0 = *(const v8h*)(qrow +  0 + lg * 16);
    const v8h c1 = *(const v8h*)(qrow +  8 + lg * 16);
    const v8h c2 = *(const v8h*)(qrow + 32 + lg * 16);
    const v8h c3 = *(const v8h*)(qrow + 40 + lg * 16);
#pragma unroll
    for (int i = 0; i < 8; i++) {
      a0[i] = c0[i]; a0[8 + i] = c1[i];
      a1[i] = c2[i]; a1[8 + i] = c3[i];
    }
  }

  const v8f zf = {0.f, 0.f, 0.f, 0.f, 0.f, 0.f, 0.f, 0.f};
  for (int kb = w; kb < 64; kb += 4) {
    const int s0 = kb * 16;
    const _Float16* krow = Kh + (((size_t)bh * CT) + s0 + ln) * CD;
    const v16h b0 = *(const v16h*)(krow +  0 + lg * 16);
    const v16h b1 = *(const v16h*)(krow + 32 + lg * 16);
    v8f c = zf;
    c = __builtin_amdgcn_wmma_f32_16x16x32_f16(false, a0, false, b0, (short)0, c, false, false);
    c = __builtin_amdgcn_wmma_f32_16x16x32_f16(false, a1, false, b1, (short)0, c, false, false);
    const int sg = s0 + ln;
#pragma unroll
    for (int e = 0; e < 8; e++) {
      const int r = e + lg * 8;
      const int tg = tBase + r;
      sc[r][sg] = (sg <= tg) ? c[e] : NEGF;
    }
  }
  __syncthreads();

  for (int r = w * 4; r < w * 4 + 4; r++) {
    float m = NEGF;
    for (int s = l; s < 1024; s += 32) m = fmaxf(m, sc[r][s]);
    m = waveMax32(m);
    float sum = 0.f;
    for (int s = l; s < 1024; s += 32) {
      const float e = __expf(sc[r][s] - m);
      sc[r][s] = e;
      sum += e;
    }
    sum = waveSum32(sum);
    const float inv = 1.f / sum;
    for (int s = l; s < 1024; s += 32) pr[r][s] = (_Float16)(sc[r][s] * inv);
  }
  __syncthreads();

  if (PASS1) {
    for (int s = tid; s < 1024; s += 128) {
      float t = 0.f;
#pragma unroll
      for (int r = 0; r < 16; r++) t += (float)pr[r][s];
      atomicAdd(&acc[s], t);
    }
    return;
  }

  const int dBase = w * 16;
  v8f o = zf;
  const _Float16* vt = VT + (((size_t)bh * CD) + dBase + ln) * CT;
  const _Float16* prow = &pr[ln][0];
  for (int c0 = 0; c0 < 1024; c0 += 32) {
    const v8h lo = *(const v8h*)(prow + c0 +     lg * 16);
    const v8h hi = *(const v8h*)(prow + c0 + 8 + lg * 16);
    v16h a;
#pragma unroll
    for (int i = 0; i < 8; i++) { a[i] = lo[i]; a[8 + i] = hi[i]; }
    const v16h b = *(const v16h*)(vt + c0 + lg * 16);
    o = __builtin_amdgcn_wmma_f32_16x16x32_f16(false, a, false, b, (short)0, o, false, false);
  }
  const int bb = bh >> 5, h = bh & 31;
#pragma unroll
  for (int e = 0; e < 8; e++) {
    const int tg = tBase + e + lg * 8;
    ctx[(((size_t)bb * CT) + tg) * CE + h * CD + dBase + ln] = o[e];
  }
}

extern "C" void kernel_launch(void* const* d_in, const int* in_sizes, int n_in,
                              void* d_out, int out_size, void* d_ws, size_t ws_size,
                              hipStream_t stream) {
  const float* hs = (const float*)d_in[0];
  const float* Wq = (const float*)d_in[2]; const float* bq = (const float*)d_in[3];
  const float* Wk = (const float*)d_in[4]; const float* bk = (const float*)d_in[5];
  const float* Wv = (const float*)d_in[6]; const float* bv = (const float*)d_in[7];
  const float* Wo = (const float*)d_in[8]; const float* bo = (const float*)d_in[9];
  float* out = (float*)d_out;

  char* ws = (char*)d_ws;
  size_t off = 0;
  auto alloc = [&](size_t bytes) -> void* {
    void* p = ws + off;
    off = (off + bytes + 255) & ~(size_t)255;
    return p;
  };

  signed char* xq  = (signed char*)alloc((size_t)CBT * CE);
  float*       sx  = (float*)alloc(CBT * sizeof(float));
  signed char* wqq = (signed char*)alloc((size_t)CE * CE);
  float*       swq = (float*)alloc(CE * sizeof(float));
  signed char* wkq = (signed char*)alloc((size_t)CE * CE);
  float*       swk = (float*)alloc(CE * sizeof(float));
  signed char* wvq = (signed char*)alloc((size_t)CE * CE);
  float*       swv = (float*)alloc(CE * sizeof(float));
  signed char* woq = (signed char*)alloc((size_t)CE * CE);
  float*       swo = (float*)alloc(CE * sizeof(float));
  float*       yq  = (float*)alloc((size_t)CBT * CE * sizeof(float));
  float*       yk  = (float*)alloc((size_t)CBT * CE * sizeof(float));
  float*       yv  = (float*)alloc((size_t)CBT * CE * sizeof(float));
  _Float16*    qh  = (_Float16*)alloc((size_t)CBH * CT * CD * 2);
  _Float16*    kh  = (_Float16*)alloc((size_t)CBH * CT * CD * 2);
  _Float16*    kqh = (_Float16*)alloc((size_t)CBH * CT * CD * 2);
  _Float16*    vqt = (_Float16*)alloc((size_t)CBH * CD * CT * 2);
  float*       acb = (float*)alloc(CT * sizeof(float));
  int*         flg = (int*)alloc(CT * sizeof(int));
  unsigned*    vmx = (unsigned*)alloc(256);
  float*       ctx = (float*)alloc((size_t)CBT * CE * sizeof(float));
  signed char* cq  = (signed char*)alloc((size_t)CBT * CE);
  float*       scx = (float*)alloc(CBT * sizeof(float));

  k_zero<<<1, 256, 0, stream>>>(acb, flg, vmx);

  k_rowquant8<<<CBT, 256, 0, stream>>>(hs, xq, sx, CE);
  k_rowquant8<<<CE, 256, 0, stream>>>(Wq, wqq, swq, CE);
  k_rowquant8<<<CE, 256, 0, stream>>>(Wk, wkq, swk, CE);
  k_rowquant8<<<CE, 256, 0, stream>>>(Wv, wvq, swv, CE);
  k_rowquant8<<<CE, 256, 0, stream>>>(Wo, woq, swo, CE);

  const dim3 gg(CE / 128, CBT / 128);
  k_gemm_i8<<<gg, 256, 0, stream>>>(xq, sx, wqq, swq, bq, yq, CBT, CE, CE);
  k_gemm_i8<<<gg, 256, 0, stream>>>(xq, sx, wkq, swk, bk, yk, CBT, CE, CE);
  k_gemm_i8<<<gg, 256, 0, stream>>>(xq, sx, wvq, swv, bv, yv, CBT, CE, CE);

  k_post_q<<<CBT, 256, 0, stream>>>(yq, qh);
  k_post_kv<<<CBT, 256, 0, stream>>>(yk, kh);

  const dim3 ga(CT / 16, CBH);
  k_attn<true><<<ga, 128, 0, stream>>>(qh, kh, nullptr, acb, nullptr);

  k_topk<<<4, 256, 0, stream>>>(acb, flg);
  k_absmax<<<256, 256, 0, stream>>>(yv, CBT * CE, vmx);
  k_requant<<<CBT, 256, 0, stream>>>(yk, yv, flg, vmx, kqh, vqt);

  k_attn<false><<<ga, 128, 0, stream>>>(qh, kqh, vqt, nullptr, ctx);

  k_rowquant8<<<CBT, 256, 0, stream>>>(ctx, cq, scx, CE);
  k_gemm_i8<<<gg, 256, 0, stream>>>(cq, scx, woq, swo, bo, out, CBT, CE, CE);
}
